// STEMModel_82935818486500
// MI455X (gfx1250) — matrix operand order repaired and checked by reading, not compiled, not run on hardware
//
#include <hip/hip_runtime.h>
#include <hip/hip_bf16.h>

// ---------------- model constants ----------------
#define BATCH 16384
#define VOC   50000
#define EDIM  16
#define NU    7
#define NW    9
#define INDIM 256
#define H1D   512
#define H2D   256
#define H3D   128
#define G1D   128
#define G2D   64

typedef __bf16 bf16;
typedef __attribute__((ext_vector_type(16))) __bf16 bf16x16;
typedef __attribute__((ext_vector_type(8)))  __bf16 bf16x8;
typedef __attribute__((ext_vector_type(8)))  float  f32x8;

__device__ __forceinline__ bf16x16 cat8(bf16x8 lo, bf16x8 hi) {
  return __builtin_shufflevector(lo, hi, 0,1,2,3,4,5,6,7,8,9,10,11,12,13,14,15);
}

// ---------------- weight convert + transpose: W[b][K][N] f32 -> Wt[b][N][K] bf16 ----
__global__ void wconv_kernel(const float* __restrict__ W, bf16* __restrict__ Wt,
                             int K, int N, int batch) {
  int i = blockIdx.x * blockDim.x + threadIdx.x;
  int tot = batch * K * N;
  if (i >= tot) return;
  int b = i / (K * N);
  int r = i - b * K * N;
  int k = r / N;
  int n = r - k * N;
  Wt[(size_t)b * N * K + (size_t)n * K + k] = (bf16)W[i];
}

// ---------------- embedding gather: build Xe[3][B][256], Gx[2][B][256] (bf16) ------
__global__ void embed_kernel(const int* __restrict__ uidx, const int* __restrict__ widx,
                             const float* __restrict__ ush, const float* __restrict__ wsh,
                             const float* __restrict__ ut,  const float* __restrict__ wt,
                             bf16* __restrict__ Xe, bf16* __restrict__ Gx) {
  int gid = blockIdx.x * blockDim.x + threadIdx.x;   // b*16 + f
  if (gid >= BATCH * 16) return;
  int b = gid >> 4;
  int f = gid & 15;
  const float *sh, *t0, *t1;
  if (f < NU) {
    int idx = uidx[b * NU + f];
    sh = ush + ((size_t)f * VOC + idx) * EDIM;
    t0 = ut  + (((size_t)(0 * NU + f)) * VOC + idx) * EDIM;
    t1 = ut  + (((size_t)(1 * NU + f)) * VOC + idx) * EDIM;
  } else {
    int fw = f - NU;
    int idx = widx[b * NW + fw];
    sh = wsh + ((size_t)fw * VOC + idx) * EDIM;
    t0 = wt  + (((size_t)(0 * NW + fw)) * VOC + idx) * EDIM;
    t1 = wt  + (((size_t)(1 * NW + fw)) * VOC + idx) * EDIM;
  }
  size_t o  = (size_t)b * INDIM + (size_t)f * EDIM;
  const size_t PB = (size_t)BATCH * INDIM;           // one plane
  bf16x8 v[5][2];                                    // Xe0,Xe1,Xe2,Gx0,Gx1 x two halves
#pragma unroll
  for (int q = 0; q < 2; ++q) {
#pragma unroll
    for (int d = 0; d < 8; ++d) {
      int k = q * 8 + d;
      float s = sh[k], a = t0[k], c = t1[k];
      v[0][q][d] = (bf16)a;
      v[1][q][d] = (bf16)c;
      v[2][q][d] = (bf16)s;
      v[3][q][d] = (bf16)(a + s);
      v[4][q][d] = (bf16)(c + s);
    }
  }
#pragma unroll
  for (int p = 0; p < 3; ++p) {
    *reinterpret_cast<bf16x8*>(Xe + p * PB + o)     = v[p][0];
    *reinterpret_cast<bf16x8*>(Xe + p * PB + o + 8) = v[p][1];
  }
#pragma unroll
  for (int p = 0; p < 2; ++p) {
    *reinterpret_cast<bf16x8*>(Gx + p * PB + o)     = v[3 + p][0];
    *reinterpret_cast<bf16x8*>(Gx + p * PB + o + 8) = v[3 + p][1];
  }
}

// ---------------- batched bf16 WMMA GEMM: C = relu(A @ W + bias) -------------------
// A  : [z][M][K]  bf16 row-major
// Wt : [z][N][K]  bf16 (transposed weights)
// bias: [z][N]    f32
// C  : [z][M][N]  bf16
// block = 256 threads = 8 waves; block tile 256(M) x 64(N); K step 32.
// Ping-pong double-buffered LDS, manually unrolled by 2 so buffer indices are
// compile-time constants (static ds offsets, no per-iter address VALU).
// One workgroup barrier per K-step; global prefetch runs under the wmma block.
// Each wave computes a 32x64 strip (2 A-frags x 4 B-frags = 8 wmma per K-step).
// All K dims here have even K/32 trip counts (8, 16, 4).
#define BM 256
#define BN 64
#define BKS 32
#define LPITCH 40   // 32 + 8 bf16 pad: 80B rows -> 16B aligned, bank-rotating

struct Prefetch { float4 a0, a1, a2, a3, b0; };

__global__ __launch_bounds__(256) void gemm_bias_relu(
    const bf16* __restrict__ A, const bf16* __restrict__ Wt,
    const float* __restrict__ bias, bf16* __restrict__ C,
    int M, int N, int K) {
  __shared__ bf16 As[2][BM][LPITCH];               // 2 x 20 KB
  __shared__ bf16 Bs[2][BN][LPITCH];               // 2 x 5 KB

  const int z = blockIdx.z;
  A    += (size_t)z * M * K;
  Wt   += (size_t)z * N * K;
  bias += (size_t)z * N;
  C    += (size_t)z * M * N;

  const int tid  = threadIdx.x;
  const int wave = tid >> 5;                       // 0..7 -> 32-row M strip
  const int lane = tid & 31;
  const int l16  = lane & 15;
  const int hi   = lane >> 4;                      // K-half / M-half selector
  const int m0   = blockIdx.y * BM;
  const int n0   = blockIdx.x * BN;

  f32x8 acc[2][4] = {};

  const int bn = tid >> 2;                         // B coop: row 0..63
  const int bc = (tid & 3) << 3;                   // B coop: 8-half chunk
  const bf16* aRow = A + (size_t)(m0 + tid) * K;   // this thread's coop A row
  const bf16* bRow = Wt + (size_t)(n0 + bn) * K + bc;

  auto fetch = [&](int kc) {
    Prefetch p;
    const float4* src = reinterpret_cast<const float4*>(aRow + kc);
    p.a0 = src[0]; p.a1 = src[1]; p.a2 = src[2]; p.a3 = src[3];
    p.b0 = *reinterpret_cast<const float4*>(bRow + kc);
    return p;
  };
  auto stage = [&](int buf, const Prefetch& p) {   // buf is a literal at call sites
    float4* dst = reinterpret_cast<float4*>(&As[buf][tid][0]);
    dst[0] = p.a0; dst[1] = p.a1; dst[2] = p.a2; dst[3] = p.a3;
    *reinterpret_cast<float4*>(&Bs[buf][bn][bc]) = p.b0;
  };
  auto compute = [&](int buf) {                    // buf is a literal at call sites
    bf16x16 af0, af1;
    const bf16* ar0 = &As[buf][wave * 32 + l16][0] + (hi ? 8 : 0);
    const bf16* ar1 = &As[buf][wave * 32 + 16 + l16][0] + (hi ? 8 : 0);
    af0 = cat8(*reinterpret_cast<const bf16x8*>(ar0),
               *reinterpret_cast<const bf16x8*>(ar0 + 16));
    af1 = cat8(*reinterpret_cast<const bf16x8*>(ar1),
               *reinterpret_cast<const bf16x8*>(ar1 + 16));
#pragma unroll
    for (int j = 0; j < 4; ++j) {
      const bf16* br = &Bs[buf][j * 16 + l16][0] + (hi ? 8 : 0);
      bf16x16 bfrag = cat8(*reinterpret_cast<const bf16x8*>(br),
                           *reinterpret_cast<const bf16x8*>(br + 16));
      acc[0][j] = __builtin_amdgcn_wmma_f32_16x16x32_bf16(false, af0, false, bfrag, (short)0, acc[0][j], false, false);
      acc[1][j] = __builtin_amdgcn_wmma_f32_16x16x32_bf16(false, af1, false, bfrag, (short)0, acc[1][j], false, false);
    }
  };

  // prologue: stage tile 0 into buffer 0
  stage(0, fetch(0));
  __syncthreads();

  // main pipeline, 2 K-steps per trip; trip count K/64 >= 1, K/32 always even
  for (int kc = 0; kc < K; kc += 2 * BKS) {
    // ---- phase A: compute buf0, stage (kc+32) -> buf1 (always exists) ----
    Prefetch p = fetch(kc + BKS);
    compute(0);
    stage(1, p);
    __syncthreads();

    // ---- phase B: compute buf1, stage (kc+64) -> buf0 (if it exists) ----
    const bool hasNext = (kc + 2 * BKS) < K;       // scalar-uniform
    if (hasNext) p = fetch(kc + 2 * BKS);
    compute(1);
    if (hasNext) stage(0, p);
    __syncthreads();
  }

  // epilogue: bias + relu + store (C frag: vgpr i -> m = i + 8*hi, n = l16)
#pragma unroll
  for (int fi = 0; fi < 2; ++fi) {
    const int mbase = m0 + wave * 32 + fi * 16 + hi * 8;
#pragma unroll
    for (int j = 0; j < 4; ++j) {
      int n = n0 + j * 16 + l16;
      float bv = bias[n];
#pragma unroll
      for (int i = 0; i < 8; ++i) {
        float vv = acc[fi][j][i] + bv;
        vv = fmaxf(vv, 0.0f);
        C[(size_t)(mbase + i) * N + n] = (bf16)vv;
      }
    }
  }
}

// ---------------- gate head: logits (64->3) + softmax, gates[2][B][4] f32 ----------
__global__ void gate_softmax_kernel(const bf16* __restrict__ G2,
                                    const float* __restrict__ gW3,  // [3][64][3]
                                    const float* __restrict__ gb3,  // [3][3]
                                    float* __restrict__ gates) {
  int gid = blockIdx.x * blockDim.x + threadIdx.x;   // t*B + b
  if (gid >= 2 * BATCH) return;
  int t = gid / BATCH;
  int b = gid - t * BATCH;
  const bf16* x = G2 + ((size_t)t * BATCH + b) * G2D;
  float l0 = gb3[t * 3 + 0], l1 = gb3[t * 3 + 1], l2 = gb3[t * 3 + 2];
  for (int i = 0; i < G2D; ++i) {
    float xv = (float)x[i];
    const float* w = gW3 + ((size_t)t * G2D + i) * 3;
    l0 += xv * w[0];
    l1 += xv * w[1];
    l2 += xv * w[2];
  }
  float mx = fmaxf(l0, fmaxf(l1, l2));
  float e0 = __expf(l0 - mx), e1 = __expf(l1 - mx), e2 = __expf(l2 - mx);
  float inv = 1.0f / (e0 + e1 + e2);
  float* o = gates + ((size_t)t * BATCH + b) * 4;
  o[0] = e0 * inv; o[1] = e1 * inv; o[2] = e2 * inv;
}

// ---------------- expert mixing: S[t][b][d] = sum_e gate[t][b][e] * E[e][b][d] -----
__global__ void mix_kernel(const float* __restrict__ gates,
                           const bf16* __restrict__ E, bf16* __restrict__ S) {
  int gid = blockIdx.x * blockDim.x + threadIdx.x;   // 2*B*32 threads, 4 d each
  if (gid >= 2 * BATCH * 32) return;
  int t = gid / (BATCH * 32);
  int r = gid - t * (BATCH * 32);
  int b = r >> 5;
  int d4 = (r & 31) << 2;
  const float* g = gates + ((size_t)t * BATCH + b) * 4;
  float g0 = g[0], g1 = g[1], g2 = g[2];
  const size_t EP = (size_t)BATCH * H3D;
  size_t eo = (size_t)b * H3D + d4;
#pragma unroll
  for (int d = 0; d < 4; ++d) {
    float v = g0 * (float)E[0 * EP + eo + d]
            + g1 * (float)E[1 * EP + eo + d]
            + g2 * (float)E[2 * EP + eo + d];
    S[(size_t)t * EP + eo + d] = (bf16)v;
  }
}

// ---------------- tower head: y[b][t] = T2[t][b] . W3[t] + b3[t] -------------------
__global__ void tower_head_kernel(const bf16* __restrict__ T2,
                                  const float* __restrict__ tW3,  // [2][64][1]
                                  const float* __restrict__ tb3,  // [2][1]
                                  float* __restrict__ out) {
  int gid = blockIdx.x * blockDim.x + threadIdx.x;   // t*B + b
  if (gid >= 2 * BATCH) return;
  int t = gid / BATCH;
  int b = gid - t * BATCH;
  const bf16* x = T2 + ((size_t)t * BATCH + b) * G2D;
  float acc = tb3[t];
  for (int i = 0; i < G2D; ++i) acc += (float)x[i] * tW3[t * G2D + i];
  out[(size_t)b * 2 + t] = acc;
}

// ===================================================================================
extern "C" void kernel_launch(void* const* d_in, const int* in_sizes, int n_in,
                              void* d_out, int out_size, void* d_ws, size_t ws_size,
                              hipStream_t stream) {
  (void)in_sizes; (void)n_in; (void)out_size; (void)ws_size;
  // inputs, setup_inputs() dict order (params dict flattened in insertion order)
  const int*   uidx   = (const int*)  d_in[0];   // [B,7]
  const int*   widx   = (const int*)  d_in[1];   // [B,9]
  const float* ush    = (const float*)d_in[2];   // [7,V,16]
  const float* wsh    = (const float*)d_in[3];   // [9,V,16]
  const float* utask  = (const float*)d_in[4];   // [2,7,V,16]
  const float* wtask  = (const float*)d_in[5];   // [2,9,V,16]
  const float* exp_W1 = (const float*)d_in[6];   const float* exp_b1 = (const float*)d_in[7];
  const float* exp_W2 = (const float*)d_in[8];   const float* exp_b2 = (const float*)d_in[9];
  const float* exp_W3 = (const float*)d_in[10];  const float* exp_b3 = (const float*)d_in[11];
  const float* gat_W1 = (const float*)d_in[12];  const float* gat_b1 = (const float*)d_in[13];
  const float* gat_W2 = (const float*)d_in[14];  const float* gat_b2 = (const float*)d_in[15];
  const float* gat_W3 = (const float*)d_in[16];  const float* gat_b3 = (const float*)d_in[17];
  const float* tow_W1 = (const float*)d_in[18];  const float* tow_b1 = (const float*)d_in[19];
  const float* tow_W2 = (const float*)d_in[20];  const float* tow_b2 = (const float*)d_in[21];
  const float* tow_W3 = (const float*)d_in[22];  const float* tow_b3 = (const float*)d_in[23];
  float* out = (float*)d_out;                    // [B,2]

  // workspace carve-up (all 256B aligned)
  char* ws = (char*)d_ws;
  size_t off = 0;
  auto carve = [&](size_t bytes) { char* p = ws + off; off += (bytes + 255) & ~(size_t)255; return p; };
  bf16* W1t  = (bf16*)carve((size_t)3 * H1D * INDIM * 2);
  bf16* W2t  = (bf16*)carve((size_t)3 * H2D * H1D  * 2);
  bf16* W3t  = (bf16*)carve((size_t)3 * H3D * H2D  * 2);
  bf16* gW1t = (bf16*)carve((size_t)2 * G1D * INDIM * 2);
  bf16* gW2t = (bf16*)carve((size_t)2 * G2D * G1D  * 2);
  bf16* tW1t = (bf16*)carve((size_t)2 * H3D * H3D  * 2);
  bf16* tW2t = (bf16*)carve((size_t)2 * G2D * H3D  * 2);
  bf16* Xe   = (bf16*)carve((size_t)3 * BATCH * INDIM * 2);
  bf16* Gx   = (bf16*)carve((size_t)2 * BATCH * INDIM * 2);
  bf16* Hb1  = (bf16*)carve((size_t)3 * BATCH * H1D * 2);
  bf16* Hb2  = (bf16*)carve((size_t)3 * BATCH * H2D * 2);
  bf16* Eo   = (bf16*)carve((size_t)3 * BATCH * H3D * 2);
  bf16* G1o  = (bf16*)carve((size_t)2 * BATCH * G1D * 2);
  bf16* G2o  = (bf16*)carve((size_t)2 * BATCH * G2D * 2);
  float* gts = (float*)carve((size_t)2 * BATCH * 4 * 4);
  bf16* So   = (bf16*)carve((size_t)2 * BATCH * H3D * 2);
  bf16* T1o  = (bf16*)carve((size_t)2 * BATCH * H3D * 2);
  bf16* T2o  = (bf16*)carve((size_t)2 * BATCH * G2D * 2);

  const int TB = 256;
  auto cdiv = [](int a, int b) { return (a + b - 1) / b; };

  // 1) weight convert+transpose (f32 -> bf16 [N][K]); ~1.0M elements total
  wconv_kernel<<<cdiv(3*INDIM*H1D, TB), TB, 0, stream>>>(exp_W1, W1t, INDIM, H1D, 3);
  wconv_kernel<<<cdiv(3*H1D*H2D,  TB), TB, 0, stream>>>(exp_W2, W2t, H1D,  H2D, 3);
  wconv_kernel<<<cdiv(3*H2D*H3D,  TB), TB, 0, stream>>>(exp_W3, W3t, H2D,  H3D, 3);
  wconv_kernel<<<cdiv(2*INDIM*G1D,TB), TB, 0, stream>>>(gat_W1, gW1t, INDIM, G1D, 2);
  wconv_kernel<<<cdiv(2*G1D*G2D,  TB), TB, 0, stream>>>(gat_W2, gW2t, G1D,  G2D, 2);
  wconv_kernel<<<cdiv(2*H3D*H3D,  TB), TB, 0, stream>>>(tow_W1, tW1t, H3D,  H3D, 2);
  wconv_kernel<<<cdiv(2*H3D*G2D,  TB), TB, 0, stream>>>(tow_W2, tW2t, H3D,  G2D, 2);

  // 2) embedding gather -> bf16 expert/gate inputs (tables are L2-resident: 153MB < 192MB)
  embed_kernel<<<cdiv(BATCH * 16, TB), TB, 0, stream>>>(uidx, widx, ush, wsh, utask, wtask, Xe, Gx);

  // 3) gate MLP (only gates 0,1; shared gate output is discarded by the reference)
  gemm_bias_relu<<<dim3(G1D/BN, BATCH/BM, 2), TB, 0, stream>>>(Gx,  gW1t, gat_b1, G1o, BATCH, G1D, INDIM);
  gemm_bias_relu<<<dim3(G2D/BN, BATCH/BM, 2), TB, 0, stream>>>(G1o, gW2t, gat_b2, G2o, BATCH, G2D, G1D);
  gate_softmax_kernel<<<cdiv(2 * BATCH, TB), TB, 0, stream>>>(G2o, gat_W3, gat_b3, gts);

  // 4) expert MLPs (batched over 3 experts)
  gemm_bias_relu<<<dim3(H1D/BN, BATCH/BM, 3), TB, 0, stream>>>(Xe,  W1t, exp_b1, Hb1, BATCH, H1D, INDIM);
  gemm_bias_relu<<<dim3(H2D/BN, BATCH/BM, 3), TB, 0, stream>>>(Hb1, W2t, exp_b2, Hb2, BATCH, H2D, H1D);
  gemm_bias_relu<<<dim3(H3D/BN, BATCH/BM, 3), TB, 0, stream>>>(Hb2, W3t, exp_b3, Eo,  BATCH, H3D, H2D);

  // 5) gated mixture -> tower inputs
  mix_kernel<<<cdiv(2 * BATCH * 32, TB), TB, 0, stream>>>(gts, Eo, So);

  // 6) towers (batched over 2 tasks) + linear heads
  gemm_bias_relu<<<dim3(H3D/BN, BATCH/BM, 2), TB, 0, stream>>>(So,  tW1t, tow_b1, T1o, BATCH, H3D, H3D);
  gemm_bias_relu<<<dim3(G2D/BN, BATCH/BM, 2), TB, 0, stream>>>(T1o, tW2t, tow_b2, T2o, BATCH, G2D, H3D);
  tower_head_kernel<<<cdiv(2 * BATCH, TB), TB, 0, stream>>>(T2o, tow_W3, tow_b3, out);
}
